// TRLModel_1382979470228
// MI455X (gfx1250) — matrix operand order repaired and checked by reading, not compiled, not run on hardware
//
#include <hip/hip_runtime.h>
#include <hip/hip_bf16.h>
#include <stdint.h>

// ---------------------------------------------------------------------------
// TRL model fused kernel for MI455X (gfx1250), wave32 + WMMA bf16 + async LDS.
//   h1  = M1^-1 x            (tridiag solve along F=512, shared matrix)
//   h2  = relu(h1 W1^T + b1) (GEMM 32768x512x2048, bf16 WMMA, f32 acc)
//   h3  = M2^-1 h2           (tridiag solve along H=2048)
//   out = h3 W2^T + b2       (GEMM 32768x2048x512)
// Fully fused per 16-row tile; weights (4 MB bf16) stay L2-resident.
// ---------------------------------------------------------------------------

typedef __attribute__((ext_vector_type(16))) __bf16 v16bf;
typedef __attribute__((ext_vector_type(8)))  float  v8f;
typedef __attribute__((ext_vector_type(4)))  int    v4i;

#define DIM_B   8
#define DIM_N   4096
#define DIM_IN  512
#define DIM_HID 2048
#define DIM_OUT 512
#define ROWS_TOTAL (DIM_B * DIM_N)

#define ROWS    16          // rows per workgroup (one 16-row WMMA M tile)
#define HSTRIDE 2052        // padded f32 LDS stride (2052 % 64 == 4 -> bank spread)
#define NTHREADS 256        // 8 waves of 32

// ---- CDNA5 async global->LDS copy (VGPR-bypassing, ASYNCcnt-tracked) ------
// ROCm 7.2 signature: (v4i* src_global, v4i* dst_lds, imm offset, imm cpol),
// generic pointers (backend infers address spaces from provenance).
__device__ __forceinline__ void async_g2l_b128(const void* g, void* l) {
#if defined(__gfx1250__) && __has_builtin(__builtin_amdgcn_global_load_async_to_lds_b128)
    __builtin_amdgcn_global_load_async_to_lds_b128(
        (v4i*)g, (v4i*)l, /*offset=*/0, /*cpol=*/0);
#else
    *(uint4*)l = *(const uint4*)g;
#endif
}

__device__ __forceinline__ void wait_async0() {
#if defined(__gfx1250__) && __has_builtin(__builtin_amdgcn_s_wait_asynccnt)
    __builtin_amdgcn_s_wait_asynccnt(0);
#else
    __builtin_amdgcn_s_waitcnt(0);
#endif
}

// ---------------- setup: convert weights to bf16 ---------------------------
__global__ __launch_bounds__(256) void cvt_weights_kernel(
    const float* __restrict__ W1, const float* __restrict__ W2,
    __bf16* __restrict__ W1b, __bf16* __restrict__ W2b)
{
    const size_t n1 = (size_t)DIM_HID * DIM_IN;
    const size_t n2 = (size_t)DIM_OUT * DIM_HID;
    const size_t stride = (size_t)gridDim.x * blockDim.x;
    for (size_t i = blockIdx.x * (size_t)blockDim.x + threadIdx.x; i < n1; i += stride)
        W1b[i] = (__bf16)W1[i];
    for (size_t i = blockIdx.x * (size_t)blockDim.x + threadIdx.x; i < n2; i += stride)
        W2b[i] = (__bf16)W2[i];
}

// ---------------- setup: Thomas factorization of tridiag M -----------------
// M = diag(softplus(d)+2) + diag(tanh(l),-1) + diag(tanh(u),+1)
__device__ __forceinline__ float softplus_f(float x) {
    float ax = fabsf(x);
    return fmaxf(x, 0.f) + log1pf(expf(-ax));
}

__global__ void factor_kernel(const float* __restrict__ d,
                              const float* __restrict__ l,
                              const float* __restrict__ u,
                              int n,
                              float* __restrict__ lo,
                              float* __restrict__ cp,
                              float* __restrict__ invp)
{
    if (blockIdx.x != 0 || threadIdx.x != 0) return;
    float diag0 = softplus_f(d[0]) + 2.0f;
    float ip = 1.0f / diag0;
    invp[0] = ip;
    cp[0]   = tanhf(u[0]) * ip;
    lo[0]   = 0.0f;
    for (int i = 1; i < n; ++i) {
        float li = tanhf(l[i - 1]);
        lo[i] = li;
        float di = softplus_f(d[i]) + 2.0f;
        float denom = di - li * cp[i - 1];
        ip = 1.0f / denom;
        invp[i] = ip;
        cp[i] = (i < n - 1) ? tanhf(u[i]) * ip : 0.0f;
    }
}

// ---------------- fused main kernel ----------------------------------------
__global__ __launch_bounds__(NTHREADS) void fused_trl_kernel(
    const float*  __restrict__ x,
    const __bf16* __restrict__ W1b, const float* __restrict__ b1,
    const __bf16* __restrict__ W2b, const float* __restrict__ b2,
    const float* __restrict__ lo1, const float* __restrict__ cp1, const float* __restrict__ ip1,
    const float* __restrict__ lo2, const float* __restrict__ cp2, const float* __restrict__ ip2,
    float* __restrict__ out)
{
    extern __shared__ unsigned char smem[];
    float*  hs = (float*)smem;                                   // ROWS * HSTRIDE f32
    __bf16* a1 = (__bf16*)(smem + (size_t)ROWS * HSTRIDE * 4);   // ROWS * DIM_IN  bf16
    __bf16* a2 = a1 + (size_t)ROWS * DIM_IN;                     // ROWS * DIM_HID bf16

    const int tid  = threadIdx.x;
    const int wave = tid >> 5;
    const int lane = tid & 31;
    const int ln   = lane & 15;     // N (or M) index within tile
    const int hi   = lane >> 4;     // K-half owned by this lane
    const size_t row0 = (size_t)blockIdx.x * ROWS;

    // ---- Phase 0: async-stage x rows into LDS (16B chunks, no VGPR bounce) ----
    for (int idx = tid; idx < ROWS * (DIM_IN / 4); idx += NTHREADS) {
        int r = idx >> 7;             // / 128 chunks per row
        int c = idx & 127;
        async_g2l_b128(&x[(row0 + r) * DIM_IN + c * 4],
                       &hs[r * HSTRIDE + c * 4]);
    }
    // warm the L2 path for this wave's first W panels
    __builtin_prefetch(&W1b[(size_t)(wave * 64) * DIM_IN], 0, 3);
    __builtin_prefetch(&W2b[(size_t)(wave * 64) * DIM_HID], 0, 3);
    wait_async0();
    __syncthreads();

    // ---- Phase 1: tridiag solve along IN (Thomas, one thread per row) ----
    if (tid < ROWS) {
        float* rp = &hs[tid * HSTRIDE];
        float prev = rp[0] * ip1[0];
        rp[0] = prev;
        for (int i = 1; i < DIM_IN; ++i) {
            prev = (rp[i] - lo1[i] * prev) * ip1[i];
            rp[i] = prev;
        }
        a1[tid * DIM_IN + (DIM_IN - 1)] = (__bf16)prev;
        for (int i = DIM_IN - 2; i >= 0; --i) {
            prev = rp[i] - cp1[i] * prev;
            a1[tid * DIM_IN + i] = (__bf16)prev;
        }
    }
    __syncthreads();

    // ---- Phase 2: GEMM1 (16x512).(512x2048) + bias + relu -> hs ----
    // 128 N-tiles; each wave owns 4 groups of 4 adjacent tiles: A frag reused 4x.
    for (int g = 0; g < 4; ++g) {
        const int n0 = (g * 8 + wave) * 64;       // 4 adjacent 16-col tiles
        v8f acc[4] = {};
        for (int k0 = 0; k0 < DIM_IN; k0 += 32) {
            v16bf af;
            // A fragment (contiguous k order, same as B fragment):
            //   lanes 0-15: K k0+0..15 ; lanes 16-31: K k0+16..31
            const __bf16* ap = &a1[ln * DIM_IN + k0 + hi * 16];
#pragma unroll
            for (int i = 0; i < 8; ++i) { af[i] = ap[i]; af[8 + i] = ap[8 + i]; }
#pragma unroll
            for (int t = 0; t < 4; ++t) {
                v16bf bfg;
                // B fragment: column = W1 row (n0 + t*16 + ln), lane-half picks K 0-15/16-31
                const __bf16* bp = &W1b[(size_t)(n0 + t * 16 + ln) * DIM_IN + k0 + hi * 16];
#pragma unroll
                for (int i = 0; i < 16; ++i) bfg[i] = bp[i];
                acc[t] = __builtin_amdgcn_wmma_f32_16x16x32_bf16(
                    false, af, false, bfg, (short)0, acc[t], false, false);
            }
        }
        // epilogue: C/D layout VGPR r -> M = r + 8*hi, N = tile_n0 + ln
#pragma unroll
        for (int t = 0; t < 4; ++t) {
#pragma unroll
            for (int r = 0; r < 8; ++r) {
                int m = r + hi * 8;
                int n = n0 + t * 16 + ln;
                float v = acc[t][r] + b1[n];
                hs[m * HSTRIDE + n] = v > 0.0f ? v : 0.0f;
            }
        }
    }
    __syncthreads();

    // ---- Phase 3: tridiag solve along HID ----
    if (tid < ROWS) {
        float* rp = &hs[tid * HSTRIDE];
        float prev = rp[0] * ip2[0];
        rp[0] = prev;
        for (int i = 1; i < DIM_HID; ++i) {
            prev = (rp[i] - lo2[i] * prev) * ip2[i];
            rp[i] = prev;
        }
        a2[tid * DIM_HID + (DIM_HID - 1)] = (__bf16)prev;
        for (int i = DIM_HID - 2; i >= 0; --i) {
            prev = rp[i] - cp2[i] * prev;
            a2[tid * DIM_HID + i] = (__bf16)prev;
        }
    }
    __syncthreads();

    // ---- Phase 4: GEMM2 (16x2048).(2048x512) + bias -> out ----
    // 32 N-tiles; each wave owns one group of 4 adjacent tiles.
    {
        const int n0 = wave * 64;
        v8f acc[4] = {};
        for (int k0 = 0; k0 < DIM_HID; k0 += 32) {
            v16bf af;
            const __bf16* ap = &a2[ln * DIM_HID + k0 + hi * 16];
#pragma unroll
            for (int i = 0; i < 8; ++i) { af[i] = ap[i]; af[8 + i] = ap[8 + i]; }
#pragma unroll
            for (int t = 0; t < 4; ++t) {
                v16bf bfg;
                const __bf16* bp = &W2b[(size_t)(n0 + t * 16 + ln) * DIM_HID + k0 + hi * 16];
#pragma unroll
                for (int i = 0; i < 16; ++i) bfg[i] = bp[i];
                acc[t] = __builtin_amdgcn_wmma_f32_16x16x32_bf16(
                    false, af, false, bfg, (short)0, acc[t], false, false);
            }
        }
#pragma unroll
        for (int t = 0; t < 4; ++t) {
#pragma unroll
            for (int r = 0; r < 8; ++r) {
                int m = r + hi * 8;
                int n = n0 + t * 16 + ln;
                out[(row0 + m) * DIM_OUT + n] = acc[t][r] + b2[n];
            }
        }
    }
}

// ---------------------------------------------------------------------------
extern "C" void kernel_launch(void* const* d_in, const int* in_sizes, int n_in,
                              void* d_out, int out_size, void* d_ws, size_t ws_size,
                              hipStream_t stream) {
    (void)in_sizes; (void)n_in; (void)out_size; (void)ws_size;

    const float* x  = (const float*)d_in[0];
    const float* d1 = (const float*)d_in[1];
    const float* l1 = (const float*)d_in[2];
    const float* u1 = (const float*)d_in[3];
    const float* W1 = (const float*)d_in[4];
    const float* b1 = (const float*)d_in[5];
    const float* d2 = (const float*)d_in[6];
    const float* l2 = (const float*)d_in[7];
    const float* u2 = (const float*)d_in[8];
    const float* W2 = (const float*)d_in[9];
    const float* b2 = (const float*)d_in[10];
    float* out = (float*)d_out;

    // ---- workspace carve (floats first, then bf16 weights) ----
    uint8_t* ws = (uint8_t*)d_ws;
    float* lo1 = (float*)ws;                 // 512
    float* cp1 = lo1 + DIM_IN;               // 512
    float* ip1 = cp1 + DIM_IN;               // 512
    float* lo2 = ip1 + DIM_IN;               // 2048
    float* cp2 = lo2 + DIM_HID;              // 2048
    float* ip2 = cp2 + DIM_HID;              // 2048
    __bf16* W1b = (__bf16*)(ws + 32768);                               // 2 MB
    __bf16* W2b = W1b + (size_t)DIM_HID * DIM_IN;                      // 2 MB

    // ---- setup kernels ----
    cvt_weights_kernel<<<2048, 256, 0, stream>>>(W1, W2, W1b, W2b);
    factor_kernel<<<1, 32, 0, stream>>>(d1, l1, u1, DIM_IN,  lo1, cp1, ip1);
    factor_kernel<<<1, 32, 0, stream>>>(d2, l2, u2, DIM_HID, lo2, cp2, ip2);

    // ---- fused main kernel: one WG per 16-row tile ----
    const size_t smem_bytes = (size_t)ROWS * HSTRIDE * sizeof(float)   // 131,328
                            + (size_t)ROWS * DIM_IN  * sizeof(__bf16)  //  16,384
                            + (size_t)ROWS * DIM_HID * sizeof(__bf16); //  65,536
    fused_trl_kernel<<<ROWS_TOTAL / ROWS, NTHREADS, smem_bytes, stream>>>(
        x, W1b, b1, W2b, b2, lo1, cp1, ip1, lo2, cp2, ip2, out);
}
